// model_v4_69020124447455
// MI455X (gfx1250) — compile-verified, matrix operand order checked by reading, not run on hardware
//
#include <hip/hip_runtime.h>
#include <hip/hip_bf16.h>

// ---------------------------------------------------------------------------
// Types for CDNA5 WMMA (wave32): v_wmma_f32_16x16x32_bf16
// ---------------------------------------------------------------------------
typedef __attribute__((ext_vector_type(16))) __bf16 v16bf;
typedef __attribute__((ext_vector_type(8)))  __bf16 v8bf;
typedef __attribute__((ext_vector_type(8)))  float  v8f;

__device__ __forceinline__ v8f wmma_bf16(v16bf a, v16bf b, v8f c) {
  // 8 args: (neg_a, A, neg_b, B, c_mod, C, reuse_a, reuse_b)
  return __builtin_amdgcn_wmma_f32_16x16x32_bf16(false, a, false, b,
                                                 (short)0, c, false, false);
}

// Load one 16x32 bf16 A/B fragment row-slice for this lane.
// Layout (ISA 7.12.2): lanes 0-15 hold K[0..7]+K[16..23], lanes 16-31 hold
// K[8..15]+K[24..31]; kb = (lane>=16)?8:0 selects the half.
__device__ __forceinline__ v16bf ld_frag(const __bf16* p, int kb) {
  v8bf lo = *(const v8bf*)(p + kb);
  v8bf hi = *(const v8bf*)(p + kb + 16);
  return __builtin_shufflevector(lo, hi, 0, 1, 2, 3, 4, 5, 6, 7,
                                         8, 9, 10, 11, 12, 13, 14, 15);
}

// CDNA5 async global->LDS copy, 16B per lane, tracked by ASYNCcnt
// (CDNA5 ISA; VDST = LDS byte address VGPR).
__device__ __forceinline__ void async_ld_b128(unsigned int lds_addr,
                                              unsigned long long gaddr) {
  asm volatile("global_load_async_to_lds_b128 %0, %1, off"
               :
               : "v"(lds_addr), "v"(gaddr)
               : "memory");
}
__device__ __forceinline__ void wait_asynccnt0() {
  asm volatile("s_wait_asynccnt 0x0" ::: "memory");
}

// ---------------------------------------------------------------------------
// Generic bf16 GEMM:  C[M][N] = act( A[M][K] @ W[N][K]^T + bias )
// Block tile 128x128, K-step 32. 256 threads = 8 waves (4 in M, 2 in N);
// each wave owns a 32x64 tile = 2x4 WMMA fragments.  Tiles are staged into
// double-buffered LDS with async global->LDS loads so the next tile's DMA
// overlaps the current tile's WMMAs.
// M, N multiples of 128; K multiple of 32 (true for all uses here).
// ---------------------------------------------------------------------------
__global__ __launch_bounds__(256) void gemm_bf16_kernel(
    const __bf16* __restrict__ A, const __bf16* __restrict__ W,
    const float* __restrict__ bias, __bf16* __restrict__ outb,
    float* __restrict__ outf, int M, int N, int K, int relu) {
  __shared__ __align__(16) __bf16 As[2][128 * 32];
  __shared__ __align__(16) __bf16 Bs[2][128 * 32];
  const int t = threadIdx.x;
  const int wave = t >> 5, lane = t & 31;
  const int lh = lane & 15, sel = lane >> 4, kb = sel * 8;
  const int wm = wave & 3, wn = wave >> 2;
  const int bm = blockIdx.y * 128, bn = blockIdx.x * 128;

  // This thread's two 16B chunks of each 128x32 tile.
  const int r0 = t >> 2, c0 = (t & 3) * 8;
  const int r1 = (t + 256) >> 2, c1 = ((t + 256) & 3) * 8;

  auto issue_tile = [&](int buf, int k0) {
    async_ld_b128((unsigned int)(uintptr_t)&As[buf][r0 * 32 + c0],
                  (unsigned long long)(uintptr_t)&A[(size_t)(bm + r0) * K + k0 + c0]);
    async_ld_b128((unsigned int)(uintptr_t)&As[buf][r1 * 32 + c1],
                  (unsigned long long)(uintptr_t)&A[(size_t)(bm + r1) * K + k0 + c1]);
    async_ld_b128((unsigned int)(uintptr_t)&Bs[buf][r0 * 32 + c0],
                  (unsigned long long)(uintptr_t)&W[(size_t)(bn + r0) * K + k0 + c0]);
    async_ld_b128((unsigned int)(uintptr_t)&Bs[buf][r1 * 32 + c1],
                  (unsigned long long)(uintptr_t)&W[(size_t)(bn + r1) * K + k0 + c1]);
  };

  v8f acc[2][4] = {};

  issue_tile(0, 0);
  wait_asynccnt0();
  __syncthreads();

  const int nk = K >> 5;
  for (int ki = 0; ki < nk; ++ki) {
    const int buf = ki & 1;
    if (ki + 1 < nk) issue_tile(buf ^ 1, (ki + 1) * 32);

    v16bf a0 = ld_frag(&As[buf][(wm * 32 + lh) * 32], kb);
    v16bf a1 = ld_frag(&As[buf][(wm * 32 + 16 + lh) * 32], kb);
#pragma unroll
    for (int nt = 0; nt < 4; ++nt) {
      v16bf bf = ld_frag(&Bs[buf][(wn * 64 + nt * 16 + lh) * 32], kb);
      acc[0][nt] = wmma_bf16(a0, bf, acc[0][nt]);
      acc[1][nt] = wmma_bf16(a1, bf, acc[1][nt]);
    }
    if (ki + 1 < nk) wait_asynccnt0();  // this wave's next-tile DMA done
    __syncthreads();                    // all waves' DMA + frag reads done
  }

  // Epilogue. C layout: VGPR r -> row = sel*8 + r, col = lane%16 (per tile).
#pragma unroll
  for (int mt = 0; mt < 2; ++mt)
#pragma unroll
    for (int nt = 0; nt < 4; ++nt) {
      int col = bn + wn * 64 + nt * 16 + lh;
      float bv = bias ? bias[col] : 0.f;
#pragma unroll
      for (int r = 0; r < 8; ++r) {
        int row = bm + wm * 32 + mt * 16 + sel * 8 + r;
        float v = acc[mt][nt][r] + bv;
        if (relu) v = fmaxf(v, 0.f);
        size_t o = (size_t)row * N + col;
        if (outb) outb[o] = (__bf16)v;
        if (outf) outf[o] = v;
      }
    }
}

// ---------------------------------------------------------------------------
// Flash attention, head_dim=64, bf16 in/out, fp32 online softmax.
// qkv: [T][3D] rows are tokens t = s*B + b.  One block = 1 (batch,head) x
// 64 query rows; 4 waves x 16 rows.  Key/value processed in 64-row tiles.
// K tile staged via async global->LDS; V tile transposed manually.
// Handles both global (win=seq=2048) and local windows (win=seq=512) via the
// token-base formula below.
// ---------------------------------------------------------------------------
__global__ __launch_bounds__(128) void flash_attn_kernel(
    const __bf16* __restrict__ qkv, __bf16* __restrict__ out,
    int Bsz, int NH_, int D3, int Dm, int win, int seqlen) {
  __shared__ __align__(16) __bf16 Ks[64 * 64];
  __shared__ __align__(16) __bf16 Vt[64 * 64];   // transposed: [hd][key]
  __shared__ __align__(16) __bf16 Ps[4 * 16 * 64];
  const int t = threadIdx.x;
  const int wave = t >> 5, lane = t & 31;
  const int lh = lane & 15, sel = lane >> 4, kb = sel * 8;
  const int batch = blockIdx.x;
  const int bi = batch / NH_, h = batch % NH_;
  const int base = (bi / Bsz) * win * Bsz + (bi % Bsz);
  const int colk = Dm + h * 64, colv = 2 * Dm + h * 64;
  const int q0 = blockIdx.y * 64;

  // Q fragments (16 rows x 64 hd) loaded straight from global in A layout.
  v16bf qf[2];
  {
    int s = q0 + wave * 16 + lh;
    const __bf16* qp = qkv + (size_t)(base + s * Bsz) * D3 + h * 64;
    qf[0] = ld_frag(qp, kb);
    qf[1] = ld_frag(qp + 32, kb);
  }
  v8f O[4] = {};
  float m8[8], l8[8];
#pragma unroll
  for (int r = 0; r < 8; ++r) { m8[r] = -3.0e38f; l8[r] = 0.f; }

  for (int kt = 0; kt < seqlen; kt += 64) {
    // Stage K tile (row-major, async DMA) and V tile (transposed) into LDS.
#pragma unroll
    for (int i = 0; i < 4; ++i) {
      int id = t + i * 128;
      int kr = id >> 3, c = (id & 7) * 8;
      const __bf16* src = qkv + (size_t)(base + (kt + kr) * Bsz) * D3;
      async_ld_b128((unsigned int)(uintptr_t)&Ks[kr * 64 + c],
                    (unsigned long long)(uintptr_t)(src + colk + c));
      v8bf vv = *(const v8bf*)(src + colv + c);
#pragma unroll
      for (int j = 0; j < 8; ++j) Vt[(c + j) * 64 + kr] = vv[j];
    }
    wait_asynccnt0();
    __syncthreads();

    // Scores S = (Q K^T) * 1/sqrt(64): 4 key tiles of 16, K-dim = 64.
    v8f sc[4];
#pragma unroll
    for (int nt = 0; nt < 4; ++nt) {
      v8f s = {};
#pragma unroll
      for (int ka = 0; ka < 2; ++ka) {
        v16bf kf = ld_frag(&Ks[(nt * 16 + lh) * 64 + 32 * ka], kb);
        s = wmma_bf16(qf[ka], kf, s);
      }
#pragma unroll
      for (int r = 0; r < 8; ++r) s[r] *= 0.125f;
      sc[nt] = s;
    }

    // Online softmax. Row m lives in one 16-lane half; reduce with shfl_xor.
#pragma unroll
    for (int r = 0; r < 8; ++r) {
      float mx = fmaxf(fmaxf(sc[0][r], sc[1][r]), fmaxf(sc[2][r], sc[3][r]));
#pragma unroll
      for (int msk = 1; msk <= 8; msk <<= 1)
        mx = fmaxf(mx, __shfl_xor(mx, msk, 32));
      float mnew = fmaxf(m8[r], mx);
      float al = __expf(m8[r] - mnew);
      m8[r] = mnew;
      float rs = 0.f;
#pragma unroll
      for (int nt = 0; nt < 4; ++nt) {
        float p = __expf(sc[nt][r] - mnew);
        sc[nt][r] = p;
        rs += p;
      }
#pragma unroll
      for (int msk = 1; msk <= 8; msk <<= 1) rs += __shfl_xor(rs, msk, 32);
      l8[r] = l8[r] * al + rs;
#pragma unroll
      for (int ot = 0; ot < 4; ++ot) O[ot][r] *= al;
    }

    // Re-stage P through this wave's private LDS region (C-layout -> A-layout
    // transpose). In-wave DScnt dependency only: no barrier, just wait.
    __bf16* Pw = &Ps[wave * 1024];
#pragma unroll
    for (int nt = 0; nt < 4; ++nt)
#pragma unroll
      for (int r = 0; r < 8; ++r)
        Pw[(sel * 8 + r) * 64 + nt * 16 + lh] = (__bf16)sc[nt][r];
    asm volatile("s_wait_dscnt 0" ::: "memory");
    v16bf pf[2];
    pf[0] = ld_frag(&Pw[lh * 64], kb);
    pf[1] = ld_frag(&Pw[lh * 64 + 32], kb);

    // O += P(16x64) @ V(64x64): 4 output tiles, K-dim = 64 keys.
#pragma unroll
    for (int ot = 0; ot < 4; ++ot)
#pragma unroll
      for (int ka = 0; ka < 2; ++ka) {
        v16bf vf = ld_frag(&Vt[(ot * 16 + lh) * 64 + 32 * ka], kb);
        O[ot] = wmma_bf16(pf[ka], vf, O[ot]);
      }
    __syncthreads();
  }

  // Normalize and write bf16 output rows (token-major [T][Dm]).
#pragma unroll
  for (int ot = 0; ot < 4; ++ot)
#pragma unroll
    for (int r = 0; r < 8; ++r) {
      int s = q0 + wave * 16 + sel * 8 + r;
      float v = O[ot][r] / l8[r];
      out[(size_t)(base + s * Bsz) * Dm + h * 64 + ot * 16 + lh] = (__bf16)v;
    }
}

// ---------------------------------------------------------------------------
// h = LayerNorm(h + a [+ b2]) * g + be   (D = 512, one block per token row)
// Writes fp32 master and bf16 copy for the next GEMM.
// ---------------------------------------------------------------------------
__global__ __launch_bounds__(256) void add_ln_kernel(
    float* __restrict__ h, const float* __restrict__ a,
    const float* __restrict__ b2, const float* __restrict__ g,
    const float* __restrict__ be, __bf16* __restrict__ hb) {
  __shared__ float red[256];
  const int tid = threadIdx.x;
  const size_t o = (size_t)blockIdx.x * 512;
  float v0 = h[o + tid] + a[o + tid];
  float v1 = h[o + tid + 256] + a[o + tid + 256];
  if (b2) { v0 += b2[o + tid]; v1 += b2[o + tid + 256]; }
  red[tid] = v0 + v1;
  __syncthreads();
  for (int st = 128; st > 0; st >>= 1) {
    if (tid < st) red[tid] += red[tid + st];
    __syncthreads();
  }
  float mu = red[0] * (1.f / 512.f);
  __syncthreads();
  float d0 = v0 - mu, d1 = v1 - mu;
  red[tid] = d0 * d0 + d1 * d1;
  __syncthreads();
  for (int st = 128; st > 0; st >>= 1) {
    if (tid < st) red[tid] += red[tid + st];
    __syncthreads();
  }
  float rstd = rsqrtf(red[0] * (1.f / 512.f) + 1e-5f);
  float y0 = d0 * rstd * g[tid] + be[tid];
  float y1 = d1 * rstd * g[tid + 256] + be[tid + 256];
  h[o + tid] = y0;
  h[o + tid + 256] = y1;
  hb[o + tid] = (__bf16)y0;
  hb[o + tid + 256] = (__bf16)y1;
}

// ---------------------------------------------------------------------------
// Embedding (K=14 GEMV) + sinusoidal positional encoding, output (S,B,D)
// token-major as t = s*B + b.
// ---------------------------------------------------------------------------
__global__ __launch_bounds__(256) void embed_kernel(
    const float* __restrict__ x, const float* __restrict__ Wemb,
    const float* __restrict__ bemb, float* __restrict__ hf,
    __bf16* __restrict__ hb, int Bsz, int S) {
  __shared__ float xr[14];
  const int tkn = blockIdx.x;
  const int s = tkn / Bsz, b = tkn % Bsz;
  if (threadIdx.x < 14)
    xr[threadIdx.x] = x[((size_t)b * S + s) * 14 + threadIdx.x];
  __syncthreads();
#pragma unroll
  for (int j = 0; j < 2; ++j) {
    int d = threadIdx.x + j * 256;
    float acc = bemb[d];
#pragma unroll
    for (int i = 0; i < 14; ++i) acc += xr[i] * Wemb[d * 14 + i];
    int p = d >> 1;
    float ang = (float)s * __expf(-logf(10000.f) * (float)(2 * p) / 512.f);
    acc += (d & 1) ? __cosf(ang) : __sinf(ang);
    size_t o = (size_t)tkn * 512 + d;
    hf[o] = acc;
    hb[o] = (__bf16)acc;
  }
}

// Final head: out[b] = h[S-1, b, :] . W_out + b_out     (OUT = 1)
__global__ __launch_bounds__(256) void head_kernel(
    const float* __restrict__ hf, const float* __restrict__ Wout,
    const float* __restrict__ bout, float* __restrict__ out, int Bsz, int S) {
  __shared__ float red[256];
  const int b = blockIdx.x;
  const size_t o = ((size_t)(S - 1) * Bsz + b) * 512;
  float p = 0.f;
  for (int d = threadIdx.x; d < 512; d += 256) p += hf[o + d] * Wout[d];
  red[threadIdx.x] = p;
  __syncthreads();
  for (int st = 128; st > 0; st >>= 1) {
    if (threadIdx.x < st) red[threadIdx.x] += red[threadIdx.x + st];
    __syncthreads();
  }
  if (threadIdx.x == 0) out[b] = red[0] + bout[0];
}

__global__ void cvt_kernel(const float* __restrict__ s,
                           __bf16* __restrict__ d, int n) {
  int i = blockIdx.x * 256 + threadIdx.x;
  if (i < n) d[i] = (__bf16)s[i];
}

// ---------------------------------------------------------------------------
// Host orchestration
// ---------------------------------------------------------------------------
extern "C" void kernel_launch(void* const* d_in, const int* in_sizes, int n_in,
                              void* d_out, int out_size, void* d_ws,
                              size_t ws_size, hipStream_t stream) {
  (void)in_sizes; (void)n_in; (void)out_size; (void)ws_size;
  constexpr int B = 8, S = 2048, D = 512, D3 = 3 * D, FF = 4 * D, NH = 8,
                L = 4, T = S * B, WIN = S / 4;

  char* ws = (char*)d_ws;
  size_t off = 0;
  auto alloc = [&](size_t bytes) -> char* {
    char* p = ws + off;
    off = (off + bytes + 255) & ~(size_t)255;
    return p;
  };
  float*  h_f   = (float*)alloc((size_t)T * D * 4);
  __bf16* h_b   = (__bf16*)alloc((size_t)T * D * 2);
  __bf16* big   = (__bf16*)alloc((size_t)T * FF * 2);  // qkv (TxD3) / ff1 (TxFF)
  __bf16* att   = (__bf16*)alloc((size_t)T * D * 2);
  float*  g_out = (float*)alloc((size_t)T * D * 4);    // also reused for ff2
  float*  l_out = (float*)alloc((size_t)T * D * 4);
  __bf16* wqkvg = (__bf16*)alloc((size_t)L * D3 * D * 2);
  __bf16* wog   = (__bf16*)alloc((size_t)L * D * D * 2);
  __bf16* wqkvl = (__bf16*)alloc((size_t)L * D3 * D * 2);
  __bf16* wol   = (__bf16*)alloc((size_t)L * D * D * 2);
  __bf16* w1    = (__bf16*)alloc((size_t)L * FF * D * 2);
  __bf16* w2    = (__bf16*)alloc((size_t)L * D * FF * 2);

  auto cvt = [&](const void* src, __bf16* dst, size_t n) {
    cvt_kernel<<<dim3((unsigned)((n + 255) / 256)), dim3(256), 0, stream>>>(
        (const float*)src, dst, (int)n);
  };
  cvt(d_in[3],  wqkvg, (size_t)L * D3 * D);
  cvt(d_in[5],  wog,   (size_t)L * D * D);
  cvt(d_in[7],  wqkvl, (size_t)L * D3 * D);
  cvt(d_in[9],  wol,   (size_t)L * D * D);
  cvt(d_in[11], w1,    (size_t)L * FF * D);
  cvt(d_in[13], w2,    (size_t)L * D * FF);

  embed_kernel<<<dim3(T), dim3(256), 0, stream>>>(
      (const float*)d_in[0], (const float*)d_in[1], (const float*)d_in[2],
      h_f, h_b, B, S);

  for (int i = 0; i < L; ++i) {
    const float* bqkvg = (const float*)d_in[4] + (size_t)i * D3;
    const float* bog   = (const float*)d_in[6] + (size_t)i * D;
    const float* bqkvl = (const float*)d_in[8] + (size_t)i * D3;
    const float* bol   = (const float*)d_in[10] + (size_t)i * D;
    const float* b1f   = (const float*)d_in[12] + (size_t)i * FF;
    const float* b2f   = (const float*)d_in[14] + (size_t)i * D;
    const float* g1    = (const float*)d_in[15] + (size_t)i * D;
    const float* be1   = (const float*)d_in[16] + (size_t)i * D;
    const float* g2    = (const float*)d_in[17] + (size_t)i * D;
    const float* be2   = (const float*)d_in[18] + (size_t)i * D;

    // ---- global attention branch ----
    gemm_bf16_kernel<<<dim3(D3 / 128, T / 128), dim3(256), 0, stream>>>(
        h_b, wqkvg + (size_t)i * D3 * D, bqkvg, big, nullptr, T, D3, D, 0);
    flash_attn_kernel<<<dim3(B * NH, S / 64), dim3(128), 0, stream>>>(
        big, att, B, NH, D3, D, S, S);
    gemm_bf16_kernel<<<dim3(D / 128, T / 128), dim3(256), 0, stream>>>(
        att, wog + (size_t)i * D * D, bog, nullptr, g_out, T, D, D, 0);

    // ---- local (windowed) attention branch ----
    gemm_bf16_kernel<<<dim3(D3 / 128, T / 128), dim3(256), 0, stream>>>(
        h_b, wqkvl + (size_t)i * D3 * D, bqkvl, big, nullptr, T, D3, D, 0);
    flash_attn_kernel<<<dim3((S / WIN) * B * NH, WIN / 64), dim3(128), 0,
                       stream>>>(big, att, B, NH, D3, D, WIN, WIN);
    gemm_bf16_kernel<<<dim3(D / 128, T / 128), dim3(256), 0, stream>>>(
        att, wol + (size_t)i * D * D, bol, nullptr, l_out, T, D, D, 0);

    add_ln_kernel<<<dim3(T), dim3(256), 0, stream>>>(h_f, g_out, l_out, g1,
                                                     be1, h_b);

    // ---- FFN ----
    gemm_bf16_kernel<<<dim3(FF / 128, T / 128), dim3(256), 0, stream>>>(
        h_b, w1 + (size_t)i * FF * D, b1f, big, nullptr, T, FF, D, 1);
    gemm_bf16_kernel<<<dim3(D / 128, T / 128), dim3(256), 0, stream>>>(
        big, w2 + (size_t)i * D * FF, b2f, nullptr, g_out, T, D, FF, 0);

    add_ln_kernel<<<dim3(T), dim3(256), 0, stream>>>(h_f, g_out, nullptr, g2,
                                                     be2, h_b);
  }

  head_kernel<<<dim3(B), dim3(256), 0, stream>>>(
      h_f, (const float*)d_in[19], (const float*)d_in[20], (float*)d_out, B, S);
}
